// DeformCrossAttention2D_16243566313696
// MI455X (gfx1250) — matrix operand order repaired and checked by reading, not compiled, not run on hardware
//
#include <hip/hip_runtime.h>
#include <hip/hip_bf16.h>
#include <math.h>

typedef __attribute__((ext_vector_type(16))) _Float16 v16h;
typedef __attribute__((ext_vector_type(8)))  _Float16 h8;
typedef __attribute__((ext_vector_type(2)))  _Float16 h2;
typedef __attribute__((ext_vector_type(8)))  float    v8f;
typedef __attribute__((ext_vector_type(4)))  float    f32x4;
typedef __attribute__((ext_vector_type(2)))  float    f32x2;

#define B_   4
#define T_   2048
#define D_   512
#define H_   8
#define P_   16
#define DH_  64
#define C_   512
#define HF_  64
#define WF_  64
#define HW_  (HF_*WF_)
#define RADIUS_ 0.08f

// ---------------- WMMA GEMM: C[M][N] = A[M][K] * B[K][N] (+bias) ----------------
#define BM 128
#define BN 128
#define BK 32
#define SKA 40          // padded LDS row stride (halves) to spread banks
#define NTHREADS 256

union V16 { v16h v; h8 h[2]; };

static __device__ inline h8 pack8v(f32x4 a, f32x4 b) {
  h8 r;
  r[0] = (_Float16)a.x; r[1] = (_Float16)a.y; r[2] = (_Float16)a.z; r[3] = (_Float16)a.w;
  r[4] = (_Float16)b.x; r[5] = (_Float16)b.y; r[6] = (_Float16)b.z; r[7] = (_Float16)b.w;
  return r;
}
static __device__ inline h8 pack8s(const float* t) {
  h8 r;
  #pragma unroll
  for (int j = 0; j < 8; j++) r[j] = (_Float16)t[j];
  return r;
}

// TA: float or _Float16 (f16 only with TRANSA=false). OUTH: store C as f16.
template<typename TA, bool TRANSA, bool HASBIAS, bool OUTH>
__launch_bounds__(NTHREADS)
__global__ void gemm_wmma_f16(const TA* __restrict__ A,
                              const float* __restrict__ Bm,
                              const float* __restrict__ bias,
                              void* __restrict__ Cv,
                              int M, int N, int K,
                              int lda, int ldb, int ldc,
                              long long strideA, long long strideC)
{
  __shared__ __align__(16) _Float16 As[2][BM * SKA];   // [m][k]
  __shared__ __align__(16) _Float16 Bs[2][BN * SKA];   // [n][k] (transposed tile)

  const int z = blockIdx.z;
  A += (long long)z * strideA;

  const int tid   = threadIdx.x;
  const int lane  = tid & 31;
  const int wid   = tid >> 5;       // 0..7
  const int waveM = wid >> 2;       // 0..1  (2 x 64 rows)
  const int waveN = wid & 3;        // 0..3  (4 x 32 cols)
  const int hs    = lane >> 4;      // lane half-group
  const int l16   = lane & 15;

  const int blockRow = blockIdx.x * BM;
  const int blockCol = blockIdx.y * BN;

  v8f acc[4][2] = {};

  // packed f16 register staging for double buffering (2 x h8 per operand)
  h8 aPk[2], bPk[2];

  auto fetch = [&](int k0) {
    if constexpr (!TRANSA) {
      const int r = tid >> 1;                 // 0..127
      const int c = (tid & 1) * 16;           // 0 or 16
      if constexpr (sizeof(TA) == 2) {
        const _Float16* src = (const _Float16*)A + (long long)(blockRow + r) * lda + (k0 + c);
        aPk[0] = *(const h8*)(src);
        aPk[1] = *(const h8*)(src + 8);
      } else {
        const float* src = (const float*)A + (long long)(blockRow + r) * lda + (k0 + c);
        f32x4 v0 = *(const f32x4*)(src);
        f32x4 v1 = *(const f32x4*)(src + 4);
        f32x4 v2 = *(const f32x4*)(src + 8);
        f32x4 v3 = *(const f32x4*)(src + 12);
        aPk[0] = pack8v(v0, v1);
        aPk[1] = pack8v(v2, v3);
      }
    } else {                                  // A stored [K][M], lda = M-stride
      const int m    = tid & 127;             // lane-consecutive m -> coalesced
      const int kseg = (tid >> 7) * 16;       // 0 or 16
      const float* src = (const float*)A + (long long)(k0 + kseg) * lda + (blockRow + m);
      float t[16];
      #pragma unroll
      for (int j = 0; j < 16; j++) t[j] = src[(long long)j * lda];
      aPk[0] = pack8s(t);
      aPk[1] = pack8s(t + 8);
    }
    {
      const int n    = tid & 127;             // lane-consecutive n -> coalesced
      const int kseg = (tid >> 7) * 16;
      const float* src = Bm + (long long)(k0 + kseg) * ldb + (blockCol + n);
      float t[16];
      #pragma unroll
      for (int j = 0; j < 16; j++) t[j] = src[(long long)j * ldb];
      bPk[0] = pack8s(t);
      bPk[1] = pack8s(t + 8);
    }
  };

  auto stage = [&](int buf) {
    if constexpr (!TRANSA) {
      const int r = tid >> 1;
      const int c = (tid & 1) * 16;
      _Float16* d = &As[buf][r * SKA + c];
      *(h8*)(d)     = aPk[0];
      *(h8*)(d + 8) = aPk[1];
    } else {
      const int m    = tid & 127;
      const int kseg = (tid >> 7) * 16;
      _Float16* d = &As[buf][m * SKA + kseg];
      *(h8*)(d)     = aPk[0];
      *(h8*)(d + 8) = aPk[1];
    }
    {
      const int n    = tid & 127;
      const int kseg = (tid >> 7) * 16;
      _Float16* d = &Bs[buf][n * SKA + kseg];
      *(h8*)(d)     = bPk[0];
      *(h8*)(d + 8) = bPk[1];
    }
  };

  const int nIter = K / BK;
  fetch(0);
  stage(0);
  __syncthreads();

  int buf = 0;
  for (int it = 0; it < nIter; it++) {
    if (it + 1 < nIter) fetch((it + 1) * BK);

    // ---- fragments per ISA layout
    V16 afrag[4];
    #pragma unroll
    for (int ms = 0; ms < 4; ms++) {
      int m = waveM * 64 + ms * 16 + l16;
      const _Float16* ap = &As[buf][m * SKA + 8 * hs];  // K = 8*hs..8*hs+7
      afrag[ms].h[0] = *(const h8*)(ap);
      afrag[ms].h[1] = *(const h8*)(ap + 16);           // K = 16+8*hs..23+8*hs
    }
    V16 bfrag[2];
    #pragma unroll
    for (int ns = 0; ns < 2; ns++) {
      int n = waveN * 32 + ns * 16 + l16;
      const _Float16* bp = &Bs[buf][n * SKA + 8 * hs];  // K = 8*hs..8*hs+7, 16+8*hs..23+8*hs
      bfrag[ns].h[0] = *(const h8*)(bp);
      bfrag[ns].h[1] = *(const h8*)(bp + 16);
    }
    #pragma unroll
    for (int ms = 0; ms < 4; ms++)
      #pragma unroll
      for (int ns = 0; ns < 2; ns++)
        acc[ms][ns] = __builtin_amdgcn_wmma_f32_16x16x32_f16(
            false, afrag[ms].v, false, bfrag[ns].v,
            (short)0, acc[ms][ns], false, false);

    if (it + 1 < nIter) stage(buf ^ 1);
    __syncthreads();
    buf ^= 1;
  }

  // ---- store (C/D layout: col = lane%16, row = r + 8*(lane/16))
  #pragma unroll
  for (int ms = 0; ms < 4; ms++) {
    #pragma unroll
    for (int ns = 0; ns < 2; ns++) {
      int ng    = blockCol + waveN * 32 + ns * 16 + l16;
      int mbase = blockRow + waveM * 64 + ms * 16 + 8 * hs;
      float bv = 0.0f;
      if constexpr (HASBIAS) bv = bias[ng];
      #pragma unroll
      for (int r = 0; r < 8; r++) {
        float val = acc[ms][ns][r];
        if constexpr (HASBIAS) val += bv;
        if constexpr (OUTH) {
          _Float16* C = (_Float16*)Cv + (long long)z * strideC;
          C[(long long)(mbase + r) * ldc + ng] = (_Float16)val;
        } else {
          float* C = (float*)Cv + (long long)z * strideC;
          C[(long long)(mbase + r) * ldc + ng] = val;
        }
      }
    }
  }
}

// ---------------- softmax + bilinear gather-accumulate ----------------
// one wave per (b, t, h); lanes 0..15 own point p; all 32 lanes accumulate
// 2 channels each. v is f16 [b][hw][d]: one head's 64 channels = 128B burst
// per corner per wave; 16MB tensor stays resident in the 192MB L2.
__launch_bounds__(256)
__global__ void deform_sample_kernel(const float* __restrict__ ref_xy,
                                     const float* __restrict__ off,
                                     const float* __restrict__ score,
                                     const _Float16* __restrict__ v,
                                     _Float16* __restrict__ ctx)
{
  const int lane = threadIdx.x & 31;
  const int wid  = (blockIdx.x << 3) + (threadIdx.x >> 5);  // 0..B*T*H-1
  const int h    = wid & (H_ - 1);
  const int bt   = wid >> 3;            // b*T + t
  const int b    = bt >> 11;            // T = 2048

  const int p = lane & 15;
  f32x2 o = *(const f32x2*)(off + (long long)bt * (H_ * P_ * 2) + h * (P_ * 2) + 2 * p);
  float s = score[(long long)bt * (H_ * P_) + h * P_ + p];

  // softmax over the 16 points (xor masks <= 8 stay inside the 16-lane group)
  float mx = s;
  mx = fmaxf(mx, __shfl_xor(mx, 1));
  mx = fmaxf(mx, __shfl_xor(mx, 2));
  mx = fmaxf(mx, __shfl_xor(mx, 4));
  mx = fmaxf(mx, __shfl_xor(mx, 8));
  float e = __expf(s - mx);
  float sum = e;
  sum += __shfl_xor(sum, 1);
  sum += __shfl_xor(sum, 2);
  sum += __shfl_xor(sum, 4);
  sum += __shfl_xor(sum, 8);
  float a = e / sum;

  float rx = ref_xy[(long long)bt * 2 + 0];
  float ry = ref_xy[(long long)bt * 2 + 1];
  // ((samp*2-1)+1)*0.5*(W-1) == samp*(W-1)
  float ix = (rx + RADIUS_ * o.x) * (float)(WF_ - 1);
  float iy = (ry + RADIUS_ * o.y) * (float)(HF_ - 1);
  float fx0 = floorf(ix), fy0 = floorf(iy);
  float fx = ix - fx0, fy = iy - fy0;
  int x0 = (int)fx0, y0 = (int)fy0;
  int x1 = x0 + 1,  y1 = y0 + 1;

  float wx0 = 1.0f - fx, wy0 = 1.0f - fy;
  float wc[4];
  int   idxc[4];
  {
    int   xs[4] = { x0, x1, x0, x1 };
    int   ys[4] = { y0, y0, y1, y1 };
    float ws[4] = { wx0 * wy0, fx * wy0, wx0 * fy, fx * fy };
    #pragma unroll
    for (int c = 0; c < 4; c++) {
      bool valid = (xs[c] >= 0) && (xs[c] < WF_) && (ys[c] >= 0) && (ys[c] < HF_);
      int xc = min(max(xs[c], 0), WF_ - 1);
      int yc = min(max(ys[c], 0), HF_ - 1);
      idxc[c] = yc * WF_ + xc;
      wc[c]   = valid ? (ws[c] * a) : 0.0f;
    }
  }

  const int ch = lane * 2;  // 2 channels per lane (64 total)
  const _Float16* vb = v + (long long)b * HW_ * D_ + h * DH_ + ch;
  float acc0 = 0.0f, acc1 = 0.0f;
  #pragma unroll
  for (int pp = 0; pp < 16; pp++) {
    #pragma unroll
    for (int c = 0; c < 4; c++) {
      int   idx = __shfl(idxc[c], pp);
      float w   = __shfl(wc[c],   pp);
      h2 vv = *(const h2*)(vb + (long long)idx * D_);
      acc0 += w * (float)vv.x;
      acc1 += w * (float)vv.y;
    }
  }
  h2 outv;
  outv.x = (_Float16)acc0;
  outv.y = (_Float16)acc1;
  *(h2*)(ctx + (long long)bt * D_ + h * DH_ + ch) = outv;
}

// ---------------- host-side launch ----------------
extern "C" void kernel_launch(void* const* d_in, const int* in_sizes, int n_in,
                              void* d_out, int out_size, void* d_ws, size_t ws_size,
                              hipStream_t stream)
{
  (void)in_sizes; (void)n_in; (void)out_size; (void)ws_size;
  const float* q      = (const float*)d_in[0];  // (B,T,D)
  const float* fmap   = (const float*)d_in[1];  // (B,C,HF,WF)
  const float* ref_xy = (const float*)d_in[2];  // (B,T,2)
  const float* Wv     = (const float*)d_in[3];  // (C,D)
  const float* W_off  = (const float*)d_in[4];  // (D,256)
  const float* b_off  = (const float*)d_in[5];
  const float* W_w    = (const float*)d_in[6];  // (D,128)
  const float* b_w    = (const float*)d_in[7];
  const float* W_out  = (const float*)d_in[8];  // (D,D)
  const float* b_out  = (const float*)d_in[9];
  float* out = (float*)d_out;

  // workspace: v (f16, 16MB) | off (f32, 8MB) | score (f32, 4MB) | ctx (f16, 8MB)
  _Float16* v_ws   = (_Float16*)d_ws;
  float* off_ws    = (float*)(v_ws + (size_t)B_ * HW_ * D_);
  float* score_ws  = off_ws + (size_t)B_ * T_ * H_ * P_ * 2;
  _Float16* ctx_ws = (_Float16*)(score_ws + (size_t)B_ * T_ * H_ * P_);

  dim3 blk(NTHREADS);

  // 1) v-proj per batch: v[hw][d] = sum_c fmap[c][hw] * Wv[c][d]  (A transposed, f16 out)
  gemm_wmma_f16<float, true, false, true><<<dim3(HW_ / BM, D_ / BN, B_), blk, 0, stream>>>(
      fmap, Wv, nullptr, v_ws, HW_, D_, C_, HW_, D_, D_,
      (long long)C_ * HW_, (long long)HW_ * D_);

  // 2) sampling offsets: q @ W_off + b_off  (f32 out)
  gemm_wmma_f16<float, false, true, false><<<dim3(B_ * T_ / BM, (H_ * P_ * 2) / BN, 1), blk, 0, stream>>>(
      q, W_off, b_off, off_ws, B_ * T_, H_ * P_ * 2, D_,
      D_, H_ * P_ * 2, H_ * P_ * 2, 0, 0);

  // 3) attention scores: q @ W_w + b_w  (f32 out)
  gemm_wmma_f16<float, false, true, false><<<dim3(B_ * T_ / BM, (H_ * P_) / BN, 1), blk, 0, stream>>>(
      q, W_w, b_w, score_ws, B_ * T_, H_ * P_, D_,
      D_, H_ * P_, H_ * P_, 0, 0);

  // 4) softmax + bilinear gather-accumulate -> ctx (f16)
  deform_sample_kernel<<<dim3(B_ * T_ * H_ / 8), dim3(256), 0, stream>>>(
      ref_xy, off_ws, score_ws, v_ws, ctx_ws);

  // 5) output projection: ctx(f16) @ W_out + b_out -> d_out (f32)
  gemm_wmma_f16<_Float16, false, true, false><<<dim3(B_ * T_ / BM, D_ / BN, 1), blk, 0, stream>>>(
      ctx_ws, W_out, b_out, out, B_ * T_, D_, D_,
      D_, D_, D_, 0, 0);
}
